// DASR_5712306504091
// MI455X (gfx1250) — compile-verified, matrix operand order checked by reading, not run on hardware
//
#include <hip/hip_runtime.h>
#include <hip/hip_bf16.h>
#include <math.h>

// ---------------------------------------------------------------------------
// DASR forward on gfx1250 (MI455X).  3x3 and 1x1 convs are implicit GEMMs on
// v_wmma_f32_16x16x32_f16.  Weights are pre-swizzled into the per-lane WMMA
// A-operand layout (CDNA5 ISA, 16-bit A 16x32 table) so each lane loads
// its 16 halves as two contiguous b128 loads.  Activations staged to LDS as
// f16 in [kh][col][ci] order so B operands are two contiguous ds b128 loads.
// ---------------------------------------------------------------------------

typedef _Float16 v16h __attribute__((ext_vector_type(16)));
typedef _Float16 v8h  __attribute__((ext_vector_type(8)));
typedef float    v8f  __attribute__((ext_vector_type(8)));

#define DEV __device__ __forceinline__

DEV float lrelu_f(float v) { return v > 0.f ? v : 0.1f * v; }

// per-lane K slot within a 32-wide chunk for WMMA 16-bit A/B operands:
// lanes 0-15 hold K = {0..7, 16..23}, lanes 16-31 hold K = {8..15, 24..31}
DEV int klocOf(int s, int laneHi) {
  int v = s >> 1, p = s & 1;
  return (v & 3) * 2 + p + (v >> 2) * 16 + laneHi * 8;
}

// ---------------------------------------------------------------------------
// Weight pre-swizzle: 3x3 conv weights [64co][64ci][3][3] fp32 ->
// bank layout [(ct*18 + chunk)*32 + lane]*16 + slot halves, where
// chunk = (kh*3+kw)*2 + cihalf, k-order = (kh*3+kw)*64 + ci.
// ---------------------------------------------------------------------------
__global__ void wconv3_cast(const float* __restrict__ src, _Float16* __restrict__ dst) {
  int blk = blockIdx.x;                 // 0..71 : ct*18 + c
  int c = blk % 18;
  int lane = threadIdx.x >> 4;          // 0..31
  int s = threadIdx.x & 15;             // 0..15
  int ct = blk / 18;
  int co = ct * 16 + (lane & 15);
  int laneHi = lane >> 4;
  int kloc = klocOf(s, laneHi);
  int kk = c >> 1, cih = c & 1;
  int ci = cih * 32 + kloc;
  int kh = kk / 3, kw = kk - kh * 3;
  float w = src[((co * 64 + ci) * 3 + kh) * 3 + kw];
  dst[(blk * 32 + lane) * 16 + s] = (_Float16)w;
}

// 1x1 conv weights [64co][64ci] fp32 -> bank (ct*2 + cihalf) blocks
__global__ void wconv1_cast(const float* __restrict__ src, _Float16* __restrict__ dst) {
  int blk = blockIdx.x;                 // 0..7 : ct*2 + c
  int c = blk & 1;
  int lane = threadIdx.x >> 4;
  int s = threadIdx.x & 15;
  int ct = blk >> 1;
  int co = ct * 16 + (lane & 15);
  int laneHi = lane >> 4;
  int ci = c * 32 + klocOf(s, laneHi);
  dst[(blk * 32 + lane) * 16 + s] = (_Float16)src[co * 64 + ci];
}

// ---------------------------------------------------------------------------
// kv = lrelu(k_v[16,256] @ comp_w[64,256]^T)
// ---------------------------------------------------------------------------
__global__ void kv_comp(const float* __restrict__ kvsrc, const float* __restrict__ cw,
                        float* __restrict__ kv) {
  int b = blockIdx.x, c = threadIdx.x;
  float s = 0.f;
  for (int m = 0; m < 256; ++m) s += kvsrc[b * 256 + m] * cw[c * 256 + m];
  kv[b * 64 + c] = lrelu_f(s);
}

// ---------------------------------------------------------------------------
// Head conv: 3->64 channels, 3x3, pad 1.  grid (b,co,h) x 64 threads (w).
// ---------------------------------------------------------------------------
__global__ void head_conv(const float* __restrict__ x, const float* __restrict__ hw,
                          const float* __restrict__ hb, float* __restrict__ out) {
  int blk = blockIdx.x;
  int h = blk & 63; int t = blk >> 6; int co = t & 63; int b = t >> 6;
  int w = threadIdx.x;
  float acc = hb[co];
  for (int ci = 0; ci < 3; ++ci)
    for (int i = 0; i < 3; ++i) {
      int hh = h + i - 1; if ((unsigned)hh >= 64u) continue;
      for (int j = 0; j < 3; ++j) {
        int ww = w + j - 1; if ((unsigned)ww >= 64u) continue;
        acc += hw[((co * 3 + ci) * 3 + i) * 3 + j] * x[((b * 3 + ci) * 64 + hh) * 64 + ww];
      }
    }
  out[((b * 64 + co) * 64 + h) * 64 + w] = acc;
}

// ---------------------------------------------------------------------------
// Per-da_conv prep: dynamic depthwise kernel ker[b,c,9] and CA gate att[b,c]
// ---------------------------------------------------------------------------
__global__ void prep_da(const float* __restrict__ kv, const float* __restrict__ kw1,
                        const float* __restrict__ kw2, const float* __restrict__ caw1,
                        const float* __restrict__ caw2, float* __restrict__ ker,
                        float* __restrict__ att) {
  __shared__ float kvs[64];
  __shared__ float t1[64];
  __shared__ float t2[8];
  int b = blockIdx.x, c = threadIdx.x;
  kvs[c] = kv[b * 64 + c];
  __syncthreads();
  float s = 0.f;
  for (int m = 0; m < 64; ++m) s += kvs[m] * kw1[c * 64 + m];
  t1[c] = lrelu_f(s);
  if (c < 8) {
    float s2 = 0.f;
    for (int m = 0; m < 64; ++m) s2 += kvs[m] * caw1[c * 64 + m];
    t2[c] = lrelu_f(s2);
  }
  __syncthreads();
  for (int j = 0; j < 9; ++j) {
    int q = c * 9 + j;
    float s3 = 0.f;
    for (int m = 0; m < 64; ++m) s3 += t1[m] * kw2[q * 64 + m];
    ker[(b * 64 + c) * 9 + j] = s3;
  }
  float sa = 0.f;
  for (int r = 0; r < 8; ++r) sa += t2[r] * caw2[c * 8 + r];
  att[b * 64 + c] = 1.f / (1.f + expf(-sa));
}

// ---------------------------------------------------------------------------
// Depthwise dynamic 3x3 conv + lrelu.  grid (b*64+c)*64+h blocks, 64 threads.
// ---------------------------------------------------------------------------
__global__ void dw_conv(const float* __restrict__ in, const float* __restrict__ ker,
                        float* __restrict__ out) {
  int blk = blockIdx.x;
  int h = blk & 63; int bc = blk >> 6;
  int w = threadIdx.x;
  const float* kp = ker + bc * 9;
  float k[9];
#pragma unroll
  for (int i = 0; i < 9; ++i) k[i] = kp[i];
  const float* base = in + (size_t)bc * 64 * 64;
  float acc = 0.f;
#pragma unroll
  for (int i = 0; i < 3; ++i) {
    int hh = h + i - 1; if ((unsigned)hh >= 64u) continue;
    const float* row = base + hh * 64;
#pragma unroll
    for (int j = 0; j < 3; ++j) {
      int ww = w + j - 1; if ((unsigned)ww >= 64u) continue;
      acc += k[i * 3 + j] * row[ww];
    }
  }
  out[((size_t)bc * 64 + h) * 64 + w] = lrelu_f(acc);
}

// ---------------------------------------------------------------------------
// 1x1 conv (WMMA) + bias + x*att gate + lrelu.  In-place on `in` is safe:
// full input tile is staged to LDS before the barrier; writes after.
// ---------------------------------------------------------------------------
__global__ __launch_bounds__(256) void conv1x1_wmma(
    const float* __restrict__ in, const _Float16* __restrict__ wb,
    const float* __restrict__ bias, const float* __restrict__ att,
    const float* __restrict__ xres, float* __restrict__ out) {
  __shared__ _Float16 Xs[64 * 64];      // [px][ci]
  const int bh = blockIdx.x;
  const int b = bh >> 6, h = bh & 63;
  for (int idx = threadIdx.x; idx < 64 * 64; idx += 256) {
    int ci = idx & 63;
    int px = idx >> 6;
    Xs[px * 64 + ci] = (_Float16)in[((b * 64 + ci) * 64 + h) * 64 + px];
  }
  __syncthreads();

  const int wave = threadIdx.x >> 5;
  const int lane = threadIdx.x & 31;
  const int ct = wave >> 1;
  const int pp = wave & 1;
  const int laneHi = lane >> 4;
  const int lmod = lane & 15;
  const int px0 = pp * 32 + lmod;

  v8f acc0 = {0.f, 0.f, 0.f, 0.f, 0.f, 0.f, 0.f, 0.f};
  v8f acc1 = acc0;
  const _Float16* wl = wb + (size_t)(ct * 2) * 512 + lane * 16;

#pragma unroll
  for (int c = 0; c < 2; ++c) {
    v8h alo = *(const v8h*)(wl + c * 512);
    v8h ahi = *(const v8h*)(wl + c * 512 + 8);
    v16h A;
#pragma unroll
    for (int i = 0; i < 8; ++i) { A[i] = alo[i]; A[i + 8] = ahi[i]; }

    const int cioff = c * 32 + laneHi * 8;
    const _Float16* p0 = &Xs[px0 * 64 + cioff];
    const _Float16* p1 = p0 + 16 * 64;
    v8h b0l = *(const v8h*)p0;      v8h b0h = *(const v8h*)(p0 + 16);
    v8h b1l = *(const v8h*)p1;      v8h b1h = *(const v8h*)(p1 + 16);
    v16h B0, B1;
#pragma unroll
    for (int i = 0; i < 8; ++i) { B0[i] = b0l[i]; B0[i + 8] = b0h[i]; B1[i] = b1l[i]; B1[i + 8] = b1h[i]; }

    acc0 = __builtin_amdgcn_wmma_f32_16x16x32_f16(false, A, false, B0, (short)0, acc0, false, false);
    acc1 = __builtin_amdgcn_wmma_f32_16x16x32_f16(false, A, false, B1, (short)0, acc1, false, false);
  }

#pragma unroll
  for (int v = 0; v < 8; ++v) {
    int co = ct * 16 + laneHi * 8 + v;
    float g = att[b * 64 + co];
    size_t i0 = (((size_t)b * 64 + co) * 64 + h) * 64 + px0;
    float r0 = acc0[v] + bias[co] + xres[i0] * g;
    float r1 = acc1[v] + bias[co] + xres[i0 + 16] * g;
    out[i0]      = lrelu_f(r0);
    out[i0 + 16] = lrelu_f(r1);
  }
}

// ---------------------------------------------------------------------------
// 3x3 conv (WMMA implicit GEMM).  grid (b*64+h, coGroup).  256 threads.
// K = 576 = 18 chunks of 32 in (kh,kw,ci) order.  Optional elementwise
// residual add and lrelu.  coutTotal = 64 or 256 (upsampler).
// ---------------------------------------------------------------------------
__global__ __launch_bounds__(256) void conv3_wmma(
    const float* __restrict__ in, const _Float16* __restrict__ wb,
    const float* __restrict__ bias, const float* __restrict__ resid,
    float* __restrict__ out, int coutTotal, int applyLrelu) {
  __shared__ _Float16 Xs[3 * 66 * 64];  // [kh][paddedCol][ci]
  const int bh = blockIdx.x;
  const int b = bh >> 6, h = bh & 63;
  const int cog = blockIdx.y;

  for (int idx = threadIdx.x; idx < 3 * 66 * 64; idx += 256) {
    int ci = idx & 63;
    int col = (idx >> 6) % 66;
    int r = idx / (66 * 64);
    int hh = h + r - 1;
    int wc = col - 1;
    float v = 0.f;
    if ((unsigned)hh < 64u && (unsigned)wc < 64u)
      v = in[((b * 64 + ci) * 64 + hh) * 64 + wc];
    Xs[(r * 66 + col) * 64 + ci] = (_Float16)v;
  }
  __syncthreads();

  const int wave = threadIdx.x >> 5;
  const int lane = threadIdx.x & 31;
  const int ct = wave >> 1;             // co tile within 64-channel group
  const int pp = wave & 1;              // pixel-tile pair
  const int laneHi = lane >> 4;
  const int lmod = lane & 15;
  const int px0 = pp * 32 + lmod;

  v8f acc0 = {0.f, 0.f, 0.f, 0.f, 0.f, 0.f, 0.f, 0.f};
  v8f acc1 = acc0;
  const _Float16* wl = wb + (size_t)(cog * 4 + ct) * 18 * 512 + lane * 16;

  for (int c = 0; c < 18; ++c) {
    const int kk = c >> 1, cih = c & 1;
    const int kh = kk / 3, kw = kk - kh * 3;

    v8h alo = *(const v8h*)(wl + c * 512);
    v8h ahi = *(const v8h*)(wl + c * 512 + 8);
    v16h A;
#pragma unroll
    for (int i = 0; i < 8; ++i) { A[i] = alo[i]; A[i + 8] = ahi[i]; }
    if (c + 1 < 18) __builtin_prefetch(wl + (c + 1) * 512, 0, 1);

    const int cioff = cih * 32 + laneHi * 8;
    const _Float16* p0 = &Xs[(kh * 66 + px0 + kw) * 64 + cioff];
    const _Float16* p1 = p0 + 16 * 64;
    v8h b0l = *(const v8h*)p0;      v8h b0h = *(const v8h*)(p0 + 16);
    v8h b1l = *(const v8h*)p1;      v8h b1h = *(const v8h*)(p1 + 16);
    v16h B0, B1;
#pragma unroll
    for (int i = 0; i < 8; ++i) { B0[i] = b0l[i]; B0[i + 8] = b0h[i]; B1[i] = b1l[i]; B1[i + 8] = b1h[i]; }

    acc0 = __builtin_amdgcn_wmma_f32_16x16x32_f16(false, A, false, B0, (short)0, acc0, false, false);
    acc1 = __builtin_amdgcn_wmma_f32_16x16x32_f16(false, A, false, B1, (short)0, acc1, false, false);
  }

#pragma unroll
  for (int v = 0; v < 8; ++v) {
    int col = ct * 16 + laneHi * 8 + v;     // channel within this 64-group
    int co = cog * 64 + col;                // global output channel
    float r0 = acc0[v] + bias[co];
    float r1 = acc1[v] + bias[co];
    if (resid) {
      size_t ri = (((size_t)b * 64 + col) * 64 + h) * 64 + px0;
      r0 += resid[ri];
      r1 += resid[ri + 16];
    }
    if (applyLrelu) { r0 = lrelu_f(r0); r1 = lrelu_f(r1); }
    size_t oi = (((size_t)b * coutTotal + co) * 64 + h) * 64 + px0;
    out[oi]      = r0;
    out[oi + 16] = r1;
  }
}

// ---------------------------------------------------------------------------
// Tail conv: pixel-shuffle(up, 2) fused into 3x3 conv 64->3 over 128x128.
// ---------------------------------------------------------------------------
__global__ void tail_conv(const float* __restrict__ up, const float* __restrict__ tw,
                          const float* __restrict__ tb, float* __restrict__ out) {
  int blk = blockIdx.x;                 // b*3*128 + o*128 + y
  int y = blk & 127; int bo = blk >> 7; int o = bo % 3; int b = bo / 3;
  int x = threadIdx.x;
  float acc = tb[o];
  for (int c = 0; c < 64; ++c) {
#pragma unroll
    for (int i = 0; i < 3; ++i) {
      int yy = y + i - 1; if ((unsigned)yy >= 128u) continue;
#pragma unroll
      for (int j = 0; j < 3; ++j) {
        int xx = x + j - 1; if ((unsigned)xx >= 128u) continue;
        int ch = c * 4 + (yy & 1) * 2 + (xx & 1);
        float v = up[(((size_t)b * 256 + ch) * 64 + (yy >> 1)) * 64 + (xx >> 1)];
        acc += tw[((o * 64 + c) * 3 + i) * 3 + j] * v;
      }
    }
  }
  out[(((size_t)b * 3 + o) * 128 + y) * 128 + x] = acc;
}

// ---------------------------------------------------------------------------
extern "C" void kernel_launch(void* const* d_in, const int* in_sizes, int n_in,
                              void* d_out, int out_size, void* d_ws, size_t ws_size,
                              hipStream_t stream) {
  (void)in_sizes; (void)n_in; (void)out_size; (void)ws_size;
  const float* x      = (const float*)d_in[0];
  const float* k_v    = (const float*)d_in[1];
  const float* head_w = (const float*)d_in[2];
  const float* head_b = (const float*)d_in[3];
  const float* comp_w = (const float*)d_in[4];
  const float* da_kw1 = (const float*)d_in[5];
  const float* da_kw2 = (const float*)d_in[6];
  const float* da_cw  = (const float*)d_in[7];
  const float* da_cb  = (const float*)d_in[8];
  const float* ca_w1  = (const float*)d_in[9];
  const float* ca_w2  = (const float*)d_in[10];
  const float* dab_cw = (const float*)d_in[11];
  const float* dab_cb = (const float*)d_in[12];
  const float* grp_w  = (const float*)d_in[13];
  const float* grp_b  = (const float*)d_in[14];
  const float* body_w = (const float*)d_in[15];
  const float* body_b = (const float*)d_in[16];
  const float* up_w   = (const float*)d_in[17];
  const float* up_b   = (const float*)d_in[18];
  const float* tail_w = (const float*)d_in[19];
  const float* tail_b = (const float*)d_in[20];

  char* wsb = (char*)d_ws;
  size_t off = 0;
  auto carve = [&](size_t bytes) -> char* {
    char* p = wsb + off; off += (bytes + 255) & ~(size_t)255; return p;
  };
  _Float16* bank3 = (_Float16*)carve((size_t)60 * 36864 * 2);   // 60 layer64 entries
  _Float16* bank1 = (_Float16*)carve((size_t)50 * 4096 * 2);
  float* kvb  = (float*)carve(16 * 64 * 4);
  float* kerb = (float*)carve(16 * 64 * 9 * 4);
  float* attb = (float*)carve(16 * 64 * 4);
  const size_t ACT = (size_t)16 * 64 * 64 * 64;                 // elements
  float* x0  = (float*)carve(ACT * 4);
  float* rb  = (float*)carve(ACT * 4);
  float* t0  = (float*)carve(ACT * 4);
  float* t1  = (float*)carve(ACT * 4);
  float* gin = (float*)carve(ACT * 4);
  float* up  = (float*)carve(ACT * 4 * 4);                      // [16,256,64,64]

  // --- weight pre-swizzle to f16 WMMA layout ---
  for (int L = 0; L < 50; ++L)
    wconv3_cast<<<72, 512, 0, stream>>>(dab_cw + (size_t)L * 36864, bank3 + (size_t)L * 36864);
  for (int g = 0; g < 5; ++g)
    wconv3_cast<<<72, 512, 0, stream>>>(grp_w + (size_t)g * 36864, bank3 + (size_t)(50 + g) * 36864);
  wconv3_cast<<<72, 512, 0, stream>>>(body_w, bank3 + (size_t)55 * 36864);
  for (int lg = 0; lg < 4; ++lg)
    wconv3_cast<<<72, 512, 0, stream>>>(up_w + (size_t)lg * 36864, bank3 + (size_t)(56 + lg) * 36864);
  for (int L = 0; L < 50; ++L)
    wconv1_cast<<<8, 512, 0, stream>>>(da_cw + (size_t)L * 4096, bank1 + (size_t)L * 4096);

  // --- head / kv ---
  kv_comp<<<16, 64, 0, stream>>>(k_v, comp_w, kvb);
  head_conv<<<16 * 64 * 64, 64, 0, stream>>>(x, head_w, head_b, x0);
  hipMemcpyAsync(rb, x0, ACT * 4, hipMemcpyDeviceToDevice, stream);

  float* res = rb;
  float* swp = t0;
  for (int g = 0; g < 5; ++g) {
    hipMemcpyAsync(gin, res, ACT * 4, hipMemcpyDeviceToDevice, stream);
    for (int n = 0; n < 5; ++n) {
      int L0 = (g * 5 + n) * 2, L1 = L0 + 1;
      // --- first da_conv + conv1 ---
      prep_da<<<16, 64, 0, stream>>>(kvb, da_kw1 + (size_t)L0 * 4096, da_kw2 + (size_t)L0 * 36864,
                                     ca_w1 + (size_t)L0 * 512, ca_w2 + (size_t)L0 * 512, kerb, attb);
      dw_conv<<<16 * 64 * 64, 64, 0, stream>>>(res, kerb, t1);
      conv1x1_wmma<<<1024, 256, 0, stream>>>(t1, bank1 + (size_t)L0 * 4096,
                                             da_cb + (size_t)L0 * 64, attb, res, swp);
      conv3_wmma<<<dim3(1024, 1), 256, 0, stream>>>(swp, bank3 + (size_t)L0 * 36864,
                                                    dab_cb + (size_t)L0 * 64, nullptr, t1, 64, 1);
      // --- second da_conv + conv2 + block residual ---
      prep_da<<<16, 64, 0, stream>>>(kvb, da_kw1 + (size_t)L1 * 4096, da_kw2 + (size_t)L1 * 36864,
                                     ca_w1 + (size_t)L1 * 512, ca_w2 + (size_t)L1 * 512, kerb, attb);
      dw_conv<<<16 * 64 * 64, 64, 0, stream>>>(t1, kerb, swp);
      conv1x1_wmma<<<1024, 256, 0, stream>>>(swp, bank1 + (size_t)L1 * 4096,
                                             da_cb + (size_t)L1 * 64, attb, t1, swp);
      conv3_wmma<<<dim3(1024, 1), 256, 0, stream>>>(swp, bank3 + (size_t)L1 * 36864,
                                                    dab_cb + (size_t)L1 * 64, res, res, 64, 0);
    }
    // group tail conv + group residual, then rotate buffers
    conv3_wmma<<<dim3(1024, 1), 256, 0, stream>>>(res, bank3 + (size_t)(50 + g) * 36864,
                                                  grp_b + (size_t)g * 64, gin, swp, 64, 0);
    float* tmp = res; res = swp; swp = tmp;
  }

  // body conv + long skip
  conv3_wmma<<<dim3(1024, 1), 256, 0, stream>>>(res, bank3 + (size_t)55 * 36864,
                                                body_b, x0, swp, 64, 0);
  res = swp;
  // upsample conv 64 -> 256
  conv3_wmma<<<dim3(1024, 4), 256, 0, stream>>>(res, bank3 + (size_t)56 * 36864,
                                                up_b, nullptr, up, 256, 0);
  // tail conv (pixel shuffle fused)
  tail_conv<<<16 * 3 * 128, 128, 0, stream>>>(up, tail_w, tail_b, (float*)d_out);
}
